// Model_10050223472992
// MI455X (gfx1250) — matrix operand order repaired and checked by reading, not compiled, not run on hardware
//
#include <hip/hip_runtime.h>

typedef __attribute__((ext_vector_type(16))) __bf16 bf16x16;
typedef __attribute__((ext_vector_type(8)))  float  f32x8;

#define NUSER 100000
#define NREPO 50000
#define DIM   128
#define NEDGE 600000
#define NLAB  200000

// ---------------------------------------------------------------- zero
__global__ void k_zero(float* __restrict__ p, unsigned long long n) {
  unsigned long long i = (unsigned long long)blockIdx.x * blockDim.x + threadIdx.x;
  unsigned long long stride = (unsigned long long)gridDim.x * blockDim.x;
  for (; i < n; i += stride) p[i] = 0.0f;
}

// --------------------------------------------- scatter-sum + degree count
// one wave per edge: lane l handles floats [4l, 4l+4)
__global__ void k_scatter(const float* __restrict__ xsrc,
                          const int* __restrict__ gidx,   // gather index per edge
                          const int* __restrict__ sidx,   // scatter index per edge
                          float* __restrict__ agg,
                          float* __restrict__ cnt,
                          int nedge) {
  unsigned int tid  = blockIdx.x * blockDim.x + threadIdx.x;
  unsigned int e    = tid >> 5;
  unsigned int lane = tid & 31;
  if (e >= (unsigned)nedge) return;
  int g = gidx[e];
  int s = sidx[e];
  const float4* xs = (const float4*)(xsrc + (size_t)g * DIM);
  float4 v = xs[lane];
  float* ag = agg + (size_t)s * DIM + lane * 4;
  atomicAdd(ag + 0, v.x);
  atomicAdd(ag + 1, v.y);
  atomicAdd(ag + 2, v.z);
  atomicAdd(ag + 3, v.w);
  if (lane == 0) atomicAdd(cnt + s, 1.0f);
}

// ------------------------------------------------------- SAGE linear (WMMA)
// out[n, :] = opt_relu( (agg[n]/max(cnt,1)) @ wl^T + bl + xdst[n] @ wr^T )
__global__ __launch_bounds__(256) void k_sage_gemm(
    const float* __restrict__ agg, const float* __restrict__ cnt,
    const float* __restrict__ xdst,
    const float* __restrict__ wl, const float* __restrict__ bl,
    const float* __restrict__ wr, float* __restrict__ out,
    int n, int do_relu) {
  __shared__ __bf16 sWl[DIM * DIM];
  __shared__ __bf16 sWr[DIM * DIM];
  __shared__ float  sB[DIM];

  int t = threadIdx.x;
  for (int i = t; i < DIM * DIM; i += 256) {
    sWl[i] = (__bf16)wl[i];
    sWr[i] = (__bf16)wr[i];
  }
  if (t < DIM) sB[t] = bl[t];
  __syncthreads();

  int wave = t >> 5;
  int lane = t & 31;
  int hi   = lane >> 4;   // K-half selector per WMMA A/B layout
  int m    = lane & 15;   // row within 16-row tile / col within 16-col tile
  int tile = blockIdx.x * 8 + wave;
  int row  = tile * 16 + m;
  int rowc = row < n ? row : n - 1;

  float ic = 1.0f / fmaxf(cnt[rowc], 1.0f);
  const float* ar = agg  + (size_t)rowc * DIM;
  const float* xr = xdst + (size_t)rowc * DIM;

  // A fragments: lanes 0-15 hold K {0..15}, lanes 16-31 hold K {16..31}
  bf16x16 aM[4], aX[4];
#pragma unroll
  for (int kb = 0; kb < 4; ++kb) {
    int k0 = kb * 32 + hi * 16;
#pragma unroll
    for (int j = 0; j < 8; ++j) {
      aM[kb][j]     = (__bf16)(ar[k0 + j] * ic);
      aM[kb][8 + j] = (__bf16)(ar[k0 + 8 + j] * ic);
      aX[kb][j]     = (__bf16)(xr[k0 + j]);
      aX[kb][8 + j] = (__bf16)(xr[k0 + 8 + j]);
    }
  }

#pragma unroll
  for (int nt = 0; nt < 8; ++nt) {
    int ncol = nt * 16 + m;
    float bias = sB[ncol];
    f32x8 c = {bias, bias, bias, bias, bias, bias, bias, bias};
#pragma unroll
    for (int kb = 0; kb < 4; ++kb) {
      int boff = ncol * DIM + kb * 32 + hi * 16;  // 16 contiguous bf16 of W row
      bf16x16 bL = *(const bf16x16*)(&sWl[boff]);
      bf16x16 bR = *(const bf16x16*)(&sWr[boff]);
      c = __builtin_amdgcn_wmma_f32_16x16x32_bf16(false, aM[kb], false, bL,
                                                  (short)0, c, false, false);
      c = __builtin_amdgcn_wmma_f32_16x16x32_bf16(false, aX[kb], false, bR,
                                                  (short)0, c, false, false);
    }
#pragma unroll
    for (int i = 0; i < 8; ++i) {
      int rr = tile * 16 + i + hi * 8;   // C layout: vgpr i -> M=i (lanes<16) / i+8
      if (rr < n) {
        float v = c[i];
        if (do_relu) v = fmaxf(v, 0.0f);
        out[(size_t)rr * DIM + ncol] = v;
      }
    }
  }
}

// ------------------------------------------------------------ edge decoder
// out[l] = sum_o (zu[ls[l]] @ wu^T + bu)[o] * (zr[ld[l]] @ wr^T + br)[o]
__global__ __launch_bounds__(256) void k_decoder(
    const float* __restrict__ zu, const float* __restrict__ zr,
    const int* __restrict__ ls, const int* __restrict__ ld,
    const float* __restrict__ wu, const float* __restrict__ bu,
    const float* __restrict__ wrw, const float* __restrict__ brw,
    float* __restrict__ out, int nlab) {
  __shared__ __bf16 sWu[DIM * DIM];
  __shared__ __bf16 sWr[DIM * DIM];
  __shared__ float  sBu[DIM];
  __shared__ float  sBr[DIM];

  int t = threadIdx.x;
  for (int i = t; i < DIM * DIM; i += 256) {
    sWu[i] = (__bf16)wu[i];
    sWr[i] = (__bf16)wrw[i];
  }
  if (t < DIM) { sBu[t] = bu[t]; sBr[t] = brw[t]; }
  __syncthreads();

  int wave = t >> 5;
  int lane = t & 31;
  int hi   = lane >> 4;
  int m    = lane & 15;
  int tile = blockIdx.x * 8 + wave;
  int l    = tile * 16 + m;
  int lc   = l < nlab ? l : nlab - 1;
  const float* ur = zu + (size_t)ls[lc] * DIM;
  const float* rr = zr + (size_t)ld[lc] * DIM;

  bf16x16 aU[4], aR[4];
#pragma unroll
  for (int kb = 0; kb < 4; ++kb) {
    int k0 = kb * 32 + hi * 16;
#pragma unroll
    for (int j = 0; j < 8; ++j) {
      aU[kb][j]     = (__bf16)ur[k0 + j];
      aU[kb][8 + j] = (__bf16)ur[k0 + 8 + j];
      aR[kb][j]     = (__bf16)rr[k0 + j];
      aR[kb][8 + j] = (__bf16)rr[k0 + 8 + j];
    }
  }

  f32x8 acc = {0, 0, 0, 0, 0, 0, 0, 0};
#pragma unroll
  for (int nt = 0; nt < 8; ++nt) {
    int ncol = nt * 16 + m;
    float biu = sBu[ncol], bir = sBr[ncol];
    f32x8 cu = {biu, biu, biu, biu, biu, biu, biu, biu};
    f32x8 cr = {bir, bir, bir, bir, bir, bir, bir, bir};
#pragma unroll
    for (int kb = 0; kb < 4; ++kb) {
      int boff = ncol * DIM + kb * 32 + hi * 16;
      bf16x16 bU = *(const bf16x16*)(&sWu[boff]);
      bf16x16 bR = *(const bf16x16*)(&sWr[boff]);
      cu = __builtin_amdgcn_wmma_f32_16x16x32_bf16(false, aU[kb], false, bU,
                                                   (short)0, cu, false, false);
      cr = __builtin_amdgcn_wmma_f32_16x16x32_bf16(false, aR[kb], false, bR,
                                                   (short)0, cr, false, false);
    }
    acc += cu * cr;
  }

  // reduce the 16 N-columns held across each 16-lane half
#pragma unroll
  for (int i = 0; i < 8; ++i) {
    float v = acc[i];
    v += __shfl_xor(v, 1, 32);
    v += __shfl_xor(v, 2, 32);
    v += __shfl_xor(v, 4, 32);
    v += __shfl_xor(v, 8, 32);
    int ml = tile * 16 + i + hi * 8;
    if (m == 0 && ml < nlab) out[ml] = v;
  }
}

// --------------------------------------------------------------- dispatch
extern "C" void kernel_launch(void* const* d_in, const int* in_sizes, int n_in,
                              void* d_out, int out_size, void* d_ws, size_t ws_size,
                              hipStream_t stream) {
  const float* x_user   = (const float*)d_in[0];
  const float* x_repo   = (const float*)d_in[1];
  const int*   edge_src = (const int*)d_in[2];
  const int*   edge_dst = (const int*)d_in[3];
  const int*   label_src= (const int*)d_in[4];
  const int*   label_dst= (const int*)d_in[5];
  const float* c1f_wl = (const float*)d_in[6];
  const float* c1f_bl = (const float*)d_in[7];
  const float* c1f_wr = (const float*)d_in[8];
  const float* c1r_wl = (const float*)d_in[9];
  const float* c1r_bl = (const float*)d_in[10];
  const float* c1r_wr = (const float*)d_in[11];
  const float* c2f_wl = (const float*)d_in[12];
  const float* c2f_bl = (const float*)d_in[13];
  const float* c2f_wr = (const float*)d_in[14];
  const float* c2r_wl = (const float*)d_in[15];
  const float* c2r_bl = (const float*)d_in[16];
  const float* c2r_wr = (const float*)d_in[17];
  const float* dec_u_w = (const float*)d_in[18];
  const float* dec_u_b = (const float*)d_in[19];
  const float* dec_r_w = (const float*)d_in[20];
  const float* dec_r_b = (const float*)d_in[21];
  (void)in_sizes; (void)n_in; (void)out_size; (void)ws_size;

  float* ws = (float*)d_ws;
  size_t off = 0;
  float* agg_repo = ws + off; off += (size_t)NREPO * DIM;
  float* agg_user = ws + off; off += (size_t)NUSER * DIM;
  float* cnt_repo = ws + off; off += NREPO;
  float* cnt_user = ws + off; off += NUSER;
  unsigned long long zero_n = off;          // contiguous accumulation region
  float* h_repo = ws + off; off += (size_t)NREPO * DIM;
  float* h_user = ws + off; off += (size_t)NUSER * DIM;
  float* z_repo = ws + off; off += (size_t)NREPO * DIM;
  float* z_user = ws + off; off += (size_t)NUSER * DIM;

  dim3 blk(256);
  int zgrid = 2048;
  int sgrid = (NEDGE * 32 + 255) / 256;
  int grepo = (NREPO + 127) / 128;
  int guser = (NUSER + 127) / 128;
  int gdec  = (NLAB + 127) / 128;

  // ---- layer 1 ----
  k_zero<<<zgrid, blk, 0, stream>>>(ws, zero_n);
  k_scatter<<<sgrid, blk, 0, stream>>>(x_user, edge_src, edge_dst, agg_repo, cnt_repo, NEDGE);
  k_scatter<<<sgrid, blk, 0, stream>>>(x_repo, edge_dst, edge_src, agg_user, cnt_user, NEDGE);
  k_sage_gemm<<<grepo, blk, 0, stream>>>(agg_repo, cnt_repo, x_repo,
                                         c1f_wl, c1f_bl, c1f_wr, h_repo, NREPO, 1);
  k_sage_gemm<<<guser, blk, 0, stream>>>(agg_user, cnt_user, x_user,
                                         c1r_wl, c1r_bl, c1r_wr, h_user, NUSER, 1);
  // ---- layer 2 ----
  k_zero<<<zgrid, blk, 0, stream>>>(ws, zero_n);
  k_scatter<<<sgrid, blk, 0, stream>>>(h_user, edge_src, edge_dst, agg_repo, cnt_repo, NEDGE);
  k_scatter<<<sgrid, blk, 0, stream>>>(h_repo, edge_dst, edge_src, agg_user, cnt_user, NEDGE);
  k_sage_gemm<<<grepo, blk, 0, stream>>>(agg_repo, cnt_repo, h_repo,
                                         c2f_wl, c2f_bl, c2f_wr, z_repo, NREPO, 0);
  k_sage_gemm<<<guser, blk, 0, stream>>>(agg_user, cnt_user, h_user,
                                         c2r_wl, c2r_bl, c2r_wr, z_user, NUSER, 0);
  // ---- decoder ----
  k_decoder<<<gdec, blk, 0, stream>>>(z_user, z_repo, label_src, label_dst,
                                      dec_u_w, dec_u_b, dec_r_w, dec_r_b,
                                      (float*)d_out, NLAB);
}
